// MultiHeadAttention_7198365188226
// MI455X (gfx1250) — matrix operand order repaired and checked by reading, not compiled, not run on hardware
//
#include <hip/hip_runtime.h>

// ---------------------------------------------------------------------------
// MHA for MI455X (gfx1250): bf16 WMMA GEMMs + LDS-free flash attention
// (S^T formulation: softmax rows are lane-local, P^T feeds PV as a B-frag)
// ---------------------------------------------------------------------------

typedef __attribute__((ext_vector_type(16))) __bf16 v16bf;
typedef __attribute__((ext_vector_type(8)))  float  v8f;

#define DEVINL __device__ __forceinline__

// A-fragment K order within a 32-element group:
// stored order [0..31] (natural) so an A-frag is one contiguous
// 32B read per lane (half 0 -> first 16, half 1 -> second 16).
DEVINL int pos32(int k) {
  return k;
}

// round-to-nearest-even f32 -> bf16
DEVINL __bf16 f2bf(float f) {
  unsigned u = __builtin_bit_cast(unsigned, f);
  u = u + 0x7FFFu + ((u >> 16) & 1u);
  unsigned short s = (unsigned short)(u >> 16);
  __bf16 b; __builtin_memcpy(&b, &s, sizeof(b));
  return b;
}

// pack two f32 -> (bf16,bf16) in one u32 (lo in [15:0], hi in [31:16])
DEVINL unsigned pack_bf2(float lo, float hi) {
  unsigned ul = __builtin_bit_cast(unsigned, lo);
  unsigned uh = __builtin_bit_cast(unsigned, hi);
  ul = ul + 0x7FFFu + ((ul >> 16) & 1u);
  uh = uh + 0x7FFFu + ((uh >> 16) & 1u);
  return (uh & 0xFFFF0000u) | (ul >> 16);
}

// A-matrix fragment 16x32 bf16 from A-permuted storage: one 32B read per lane.
DEVINL v16bf load_a_frag_p(const __bf16* base, int ld, int lane) {
  const int m   = lane & 15;
  const int off = (lane >> 4) << 4;                  // 0 or 16
  return *(const v16bf*)(base + (size_t)m * ld + off);
}

// B-matrix fragment 32x16 bf16 from natural storage (column n = row n of base):
// lanes 0-15 hold K=0..15, lanes 16-31 hold K=16..31 (contiguous per lane).
DEVINL v16bf load_b_frag(const __bf16* base, int ld, int lane) {
  const int n    = lane & 15;
  const int koff = (lane >> 4) << 4;                 // 0 or 16
  return *(const v16bf*)(base + (size_t)n * ld + koff);
}

// ---------------------------------------------------------------------------
// fp32 -> bf16 conversion; PERM applies the A-frag K-permutation per 32-group
// ---------------------------------------------------------------------------
template <bool PERM>
__global__ void cvt_f32_bf16(const float* __restrict__ in,
                             __bf16* __restrict__ out, int n4) {
  int i = blockIdx.x * blockDim.x + threadIdx.x;
  if (i >= n4) return;
  float4 v = ((const float4*)in)[i];
  unsigned p0 = pack_bf2(v.x, v.y);
  unsigned p1 = pack_bf2(v.z, v.w);
  int f = i * 4;
  int of = PERM ? ((f & ~31) | pos32(f & 31)) : f;
  uint2 st; st.x = p0; st.y = p1;
  *(uint2*)(out + of) = st;
}

// ---------------------------------------------------------------------------
// GEMM: Y[m,n] = sum_k X[m,k]*W[n,k] + bias[n]   (M=4096, N=K=1024)
// X in A-permuted layout, W natural. 8 waves (4x2), wave tile 32x64,
// block tile 128x128.
// ---------------------------------------------------------------------------
enum OutMode { OUT_BF16_ROW = 0, OUT_BF16_KPERM = 1, OUT_BF16_VT = 2, OUT_F32_ROW = 3 };

template <int MODE>
__global__ __launch_bounds__(256)
void gemm_xwT_bf16(const __bf16* __restrict__ X,
                   const __bf16* __restrict__ W,
                   const float*  __restrict__ bias,
                   void* __restrict__ Yout) {
  constexpr int NN = 1024, KK = 1024;
  const int lane = threadIdx.x & 31;
  const int wave = threadIdx.x >> 5;
  const int m0 = blockIdx.x * 128 + (wave >> 1) * 32;
  const int n0 = blockIdx.y * 128 + (wave & 1) * 64;

  v8f acc[2][4] = {};
  const __bf16* A0 = X + (size_t)m0 * KK;
  const __bf16* A1 = X + (size_t)(m0 + 16) * KK;

  for (int k0 = 0; k0 < KK; k0 += 32) {
    v16bf a0 = load_a_frag_p(A0 + k0, KK, lane);
    v16bf a1 = load_a_frag_p(A1 + k0, KK, lane);
#pragma unroll
    for (int j = 0; j < 4; ++j) {
      v16bf b = load_b_frag(W + (size_t)(n0 + 16 * j) * KK + k0, KK, lane);
      acc[0][j] = __builtin_amdgcn_wmma_f32_16x16x32_bf16(
          false, a0, false, b, (short)0, acc[0][j], false, false);
      acc[1][j] = __builtin_amdgcn_wmma_f32_16x16x32_bf16(
          false, a1, false, b, (short)0, acc[1][j], false, false);
    }
  }

  // C/D layout: element e -> row +e+8*(lane>=16), col n0+16j+(lane&15)
  const int ncol  = lane & 15;
  const int rbase = (lane >> 4) << 3;
#pragma unroll
  for (int mi = 0; mi < 2; ++mi) {
#pragma unroll
    for (int j = 0; j < 4; ++j) {
      const int n = n0 + 16 * j + ncol;
      const float bv = bias[n];
#pragma unroll
      for (int e = 0; e < 8; ++e) {
        const int m = m0 + 16 * mi + rbase + e;
        const float val = acc[mi][j][e] + bv;
        if (MODE == OUT_F32_ROW) {
          ((float*)Yout)[(size_t)m * NN + n] = val;
        } else if (MODE == OUT_BF16_ROW) {
          ((__bf16*)Yout)[(size_t)m * NN + n] = f2bf(val);
        } else if (MODE == OUT_BF16_KPERM) {   // A-permuted along n (K dim of consumer)
          const int np = (n & ~31) | pos32(n & 31);
          ((__bf16*)Yout)[(size_t)m * NN + np] = f2bf(val);
        } else {                               // OUT_BF16_VT: Vt[b,h,d,s], s A-permuted
          const int b = m >> 11, s = m & 2047;
          const int sp = (s & ~31) | pos32(s & 31);
          ((__bf16*)Yout)[((size_t)(b * 1024 + n) << 11) + sp] = f2bf(val);
        }
      }
    }
  }
}

// ---------------------------------------------------------------------------
// Flash attention (unscaled, matching reference), S^T formulation, no LDS.
// Grid (S/64, B*H); block 128 = 4 waves; wave owns 16 query rows.
//   S^T = K·Q^T  -> C-frag: lane = query column m, elements = t rows.
//   out^T = V^T·P^T -> A-frag from Vt (contiguous), B-frag = P^T built in regs
//   with 4 packed shfl_xor(16) exchanges between lane pairs (m, m+16).
// ---------------------------------------------------------------------------
__global__ __launch_bounds__(128)
void flash_attn_bf16(const __bf16* __restrict__ Q,   // [B,S,E] natural bf16
                     const __bf16* __restrict__ Km,  // [B,S,E] A-perm bf16
                     const __bf16* __restrict__ Vt,  // [B,H,DH,S] A-perm bf16
                     __bf16* __restrict__ O) {       // [B,S,E] A-perm bf16
  constexpr int S = 2048, E = 1024;
  const int lane = threadIdx.x & 31;
  const int wave = threadIdx.x >> 5;
  const int m0 = blockIdx.x * 64 + wave * 16;
  const int bh = blockIdx.y;
  const int b = bh >> 4, h = bh & 15;
  const bool hi = lane >= 16;

  const __bf16* Qb = Q  + ((size_t)(b * S + m0)) * E + h * 64;
  const __bf16* Kb = Km + ((size_t)(b * S)) * E + h * 64;
  const __bf16* Vb = Vt + ((size_t)bh * 64) * S;

  const v16bf qb0 = load_b_frag(Qb, E, lane);        // Q^T B-frag, d = 0..31
  const v16bf qb1 = load_b_frag(Qb + 32, E, lane);   // d = 32..63

  v8f acc[4] = {};
  float mrun = -3.0e38f, lrun = 0.0f;

  for (int t0 = 0; t0 < S; t0 += 32) {
    // ---- S^T tile (32 t-rows x 16 m-cols): sA -> t0+e+8h, sB -> t0+16+e+8h
    v8f sA = {}, sB = {};
    {
      v16bf ka;
      ka = load_a_frag_p(Kb + (size_t)t0 * E, E, lane);
      sA = __builtin_amdgcn_wmma_f32_16x16x32_bf16(false, ka, false, qb0, (short)0, sA, false, false);
      ka = load_a_frag_p(Kb + (size_t)t0 * E + 32, E, lane);
      sA = __builtin_amdgcn_wmma_f32_16x16x32_bf16(false, ka, false, qb1, (short)0, sA, false, false);
      ka = load_a_frag_p(Kb + (size_t)(t0 + 16) * E, E, lane);
      sB = __builtin_amdgcn_wmma_f32_16x16x32_bf16(false, ka, false, qb0, (short)0, sB, false, false);
      ka = load_a_frag_p(Kb + (size_t)(t0 + 16) * E + 32, E, lane);
      sB = __builtin_amdgcn_wmma_f32_16x16x32_bf16(false, ka, false, qb1, (short)0, sB, false, false);
    }

    // ---- prefetch next K/V tiles (global_prefetch_b8)
    if (t0 + 32 < S) {
      __builtin_prefetch(Kb + (size_t)(t0 + 32 + lane) * E, 0, 1);
      __builtin_prefetch(Vb + (size_t)lane * S + t0 + 32, 0, 1);
      __builtin_prefetch(Vb + (size_t)(lane + 32) * S + t0 + 32, 0, 1);
    }

    // ---- online softmax: rows are lane-local; one xor-16 combines lane pair
    float mloc = fmaxf(sA[0], sB[0]);
#pragma unroll
    for (int e = 1; e < 8; ++e) mloc = fmaxf(mloc, fmaxf(sA[e], sB[e]));
    mloc = fmaxf(mloc, __shfl_xor(mloc, 16, 32));
    const float mnew = fmaxf(mrun, mloc);
    const float scl = __expf(mrun - mnew);
    mrun = mnew;

    float pA[8], pB[8], ls = 0.0f;
#pragma unroll
    for (int e = 0; e < 8; ++e) {
      pA[e] = __expf(sA[e] - mnew);
      pB[e] = __expf(sB[e] - mnew);
      ls += pA[e] + pB[e];
    }
    ls += __shfl_xor(ls, 16, 32);
    lrun = lrun * scl + ls;
#pragma unroll
    for (int o = 0; o < 4; ++o)
#pragma unroll
      for (int e = 0; e < 8; ++e) acc[o][e] *= scl;

    // ---- build P^T B-frag: lane m needs t0..t0+15, lane m+16 needs t0+16..31
    // half0 owns t0+0..7 (pA) and t0+16..23 (pB); half1 owns t0+8..15, t0+24..31.
    unsigned snd[4], rcv[4];
#pragma unroll
    for (int i = 0; i < 4; ++i)
      snd[i] = hi ? pack_bf2(pA[2 * i], pA[2 * i + 1])   // half1 sends its pA (t+8..15)
                  : pack_bf2(pB[2 * i], pB[2 * i + 1]);  // half0 sends its pB (t+16..23)
#pragma unroll
    for (int i = 0; i < 4; ++i) rcv[i] = __shfl_xor(snd[i], 16, 32);

    union { unsigned u[8]; v16bf v; } pb;
#pragma unroll
    for (int i = 0; i < 4; ++i) {
      pb.u[i]     = hi ? rcv[i] : pack_bf2(pA[2 * i], pA[2 * i + 1]);
      pb.u[4 + i] = hi ? pack_bf2(pB[2 * i], pB[2 * i + 1]) : rcv[i];
    }

    // ---- out^T += V^T · P^T  (4 d-tiles of 16)
#pragma unroll
    for (int o = 0; o < 4; ++o) {
      v16bf va = load_a_frag_p(Vb + (size_t)(o * 16) * S + t0, S, lane);
      acc[o] = __builtin_amdgcn_wmma_f32_16x16x32_bf16(
          false, va, false, pb.v, (short)0, acc[o], false, false);
    }
  }

  // ---- normalize + store: lane writes row m0+m, 8 contiguous d per frag,
  // in the natural (identity-permuted) layout of O.
  const float inv = 1.0f / lrun;
  __bf16* Ob = O + ((size_t)(b * S + m0 + (lane & 15))) * E + h * 64;
  const int rmap[4] = {0, 1, 2, 3};
#pragma unroll
  for (int o = 0; o < 4; ++o) {
    const int r = 2 * (o & 1) + (hi ? 1 : 0);
    const int off = (o >> 1) * 32 + 8 * rmap[r];
    uint4 st;
    st.x = pack_bf2(acc[o][0] * inv, acc[o][1] * inv);
    st.y = pack_bf2(acc[o][2] * inv, acc[o][3] * inv);
    st.z = pack_bf2(acc[o][4] * inv, acc[o][5] * inv);
    st.w = pack_bf2(acc[o][6] * inv, acc[o][7] * inv);
    *(uint4*)(Ob + off) = st;
  }
}

// ---------------------------------------------------------------------------
// Host-side orchestration
// ---------------------------------------------------------------------------
extern "C" void kernel_launch(void* const* d_in, const int* in_sizes, int n_in,
                              void* d_out, int out_size, void* d_ws, size_t ws_size,
                              hipStream_t stream) {
  (void)in_sizes; (void)n_in; (void)out_size; (void)ws_size;

  const float* query = (const float*)d_in[0];
  const float* key_  = (const float*)d_in[1];
  const float* value = (const float*)d_in[2];
  const float* Wq = (const float*)d_in[3];
  const float* bq = (const float*)d_in[4];
  const float* Wk = (const float*)d_in[5];
  const float* bk = (const float*)d_in[6];
  const float* Wv = (const float*)d_in[7];
  const float* bv = (const float*)d_in[8];
  const float* Wo = (const float*)d_in[9];
  const float* bo = (const float*)d_in[10];

  const size_t MB = 1u << 20;
  char* ws = (char*)d_ws;
  __bf16* XB  = (__bf16*)(ws + 0);        // 8 MB staging: x_bf (A-perm), later O_bf
  __bf16* WQB = (__bf16*)(ws + 8 * MB);   // 2 MB each, natural (B-side)
  __bf16* WKB = (__bf16*)(ws + 10 * MB);
  __bf16* WVB = (__bf16*)(ws + 12 * MB);
  __bf16* WOB = (__bf16*)(ws + 14 * MB);
  __bf16* QB  = (__bf16*)(ws + 16 * MB);  // 8 MB natural (B-side in flash)
  __bf16* KB  = (__bf16*)(ws + 24 * MB);  // 8 MB A-perm (A-side in flash)
  __bf16* VT  = (__bf16*)(ws + 32 * MB);  // 8 MB A-perm transposed [B,H,DH,S]

  const int NX = 2 * 2048 * 1024;
  const int NW = 1024 * 1024;
  const dim3 cblk(256);
  const dim3 cgW((NW / 4 + 255) / 256);
  const dim3 cgX((NX / 4 + 255) / 256);
  const dim3 gg(4096 / 128, 1024 / 128);  // 32 x 8
  const dim3 gb(256);
  const dim3 fg(2048 / 64, 2 * 16);       // 32 x 32
  const dim3 fb(128);

  cvt_f32_bf16<false><<<cgW, cblk, 0, stream>>>(Wq, WQB, NW / 4);
  cvt_f32_bf16<false><<<cgW, cblk, 0, stream>>>(Wk, WKB, NW / 4);
  cvt_f32_bf16<false><<<cgW, cblk, 0, stream>>>(Wv, WVB, NW / 4);
  cvt_f32_bf16<false><<<cgW, cblk, 0, stream>>>(Wo, WOB, NW / 4);

  cvt_f32_bf16<true><<<cgX, cblk, 0, stream>>>(query, XB, NX / 4);
  gemm_xwT_bf16<OUT_BF16_ROW><<<gg, gb, 0, stream>>>(XB, WQB, bq, QB);

  cvt_f32_bf16<true><<<cgX, cblk, 0, stream>>>(key_, XB, NX / 4);
  gemm_xwT_bf16<OUT_BF16_KPERM><<<gg, gb, 0, stream>>>(XB, WKB, bk, KB);

  cvt_f32_bf16<true><<<cgX, cblk, 0, stream>>>(value, XB, NX / 4);
  gemm_xwT_bf16<OUT_BF16_VT><<<gg, gb, 0, stream>>>(XB, WVB, bv, VT);

  flash_attn_bf16<<<fg, fb, 0, stream>>>(QB, KB, VT, XB);

  gemm_xwT_bf16<OUT_F32_ROW><<<gg, gb, 0, stream>>>(XB, WOB, bo, d_out);
}
